// CausalSelfAttention_41360535060851
// MI455X (gfx1250) — matrix operand order repaired and checked by reading, not compiled, not run on hardware
//
#include <hip/hip_runtime.h>

#define DM    2048
#define NH    16
#define HD    128
#define BATCH 4
#define SEQ   2048
#define N3    (3 * DM)
#define BHN   (BATCH * NH)    // 64
#define MROWS (BATCH * SEQ)   // 8192
#define KT    32              // keys per attention tile

typedef __bf16 bf16_t;
typedef bf16_t v16bf __attribute__((ext_vector_type(16)));
typedef float  v8f   __attribute__((ext_vector_type(8)));
typedef unsigned short u16x16 __attribute__((ext_vector_type(16)));
typedef unsigned short u16x8  __attribute__((ext_vector_type(8)));

__device__ __forceinline__ unsigned short f2bf(float f) {
  unsigned int u = __float_as_uint(f);
  u += 0x7fffu + ((u >> 16) & 1u);   // round-to-nearest-even
  return (unsigned short)(u >> 16);
}
__device__ __forceinline__ float bf2f(unsigned short h) {
  return __uint_as_float(((unsigned int)h) << 16);
}
__device__ __forceinline__ u16x16 comb8(u16x8 lo, u16x8 hs) {
  u16x16 r;
#pragma unroll
  for (int i = 0; i < 8; ++i) { r[i] = lo[i]; r[i + 8] = hs[i]; }
  return r;
}
__device__ __forceinline__ v8f wmma_bf(u16x16 a, u16x16 b, v8f c) {
  return __builtin_amdgcn_wmma_f32_16x16x32_bf16(
      false, __builtin_bit_cast(v16bf, a),
      false, __builtin_bit_cast(v16bf, b),
      (short)0, c, false, false);
}
// async copy of 16 bytes per lane: global -> LDS, tracked by ASYNCcnt
__device__ __forceinline__ void async_copy16(const unsigned short* g,
                                             const unsigned short* l) {
  unsigned loff = (unsigned)(size_t)(const void*)l;            // LDS byte offset
  unsigned long long ga = (unsigned long long)(size_t)(const void*)g;
  asm volatile("global_load_async_to_lds_b128 %0, %1, off"
               :: "v"(loff), "v"(ga) : "memory");
}

// ---------------- elementwise prep kernels ----------------

__global__ void k_f32_to_bf16(const float* __restrict__ in,
                              unsigned short* __restrict__ out, int n) {
  int i = blockIdx.x * blockDim.x + threadIdx.x;
  if (i < n) out[i] = f2bf(in[i]);
}

// W[k*N + n] (f32, KxN) -> Wt[n*K + k] (bf16, NxK), K == DM (power of two)
__global__ void k_transpose_w(const float* __restrict__ W,
                              unsigned short* __restrict__ Wt, int N, int total) {
  int i = blockIdx.x * blockDim.x + threadIdx.x;
  if (i >= total) return;
  int k = i & (DM - 1);
  int n = i >> 11;   // /DM
  Wt[i] = f2bf(W[(size_t)k * N + n]);
}

// In-place RoPE on Q and K head-major [BH][SEQ][HD] bf16 buffers.
__global__ void k_rope(unsigned short* __restrict__ Qh,
                       unsigned short* __restrict__ Kh,
                       const float* __restrict__ cosT,
                       const float* __restrict__ sinT, int total) {
  int i = blockIdx.x * blockDim.x + threadIdx.x;
  if (i >= total) return;
  const int half = BHN * SEQ * (HD / 2);
  unsigned short* buf = (i < half) ? Qh : Kh;
  int r  = (i < half) ? i : i - half;
  const int dp = r & 63;                // pair index 0..63
  const int t  = (r >> 6) & (SEQ - 1);
  const int bh = r >> 17;               // /(64*SEQ)
  size_t base = ((size_t)bh * SEQ + t) * HD;
  float x1 = bf2f(buf[base + dp]);
  float x2 = bf2f(buf[base + dp + 64]);
  float c1 = cosT[t * HD + dp],      s1 = sinT[t * HD + dp];
  float c2 = cosT[t * HD + dp + 64], s2 = sinT[t * HD + dp + 64];
  buf[base + dp]      = f2bf(x1 * c1 - x2 * s1);   // x*cos + (-x2)*sin
  buf[base + dp + 64] = f2bf(x2 * c2 + x1 * s2);   // x*cos + ( x1)*sin
}

// ---------------- WMMA GEMM core: wave computes a 16(M) x 64(N) strip ----------------
// A: bf16 row-major [*, lda];  Bt: bf16 [N][ldb] (i.e. B transposed, contiguous K per row)
__device__ __forceinline__ void gemm16x64(const unsigned short* __restrict__ A, int lda,
                                          const unsigned short* __restrict__ Bt, int ldb,
                                          int rowBase, int colBase, int K, v8f c[4]) {
  const int lane   = threadIdx.x & 31;
  const int lane16 = lane & 15;
  const int hi     = lane >> 4;
  const unsigned short* arow = A + (size_t)(rowBase + lane16) * lda;
  for (int k = 0; k < K; k += 32) {
    u16x8 alo = *(const u16x8*)(arow + k + hi * 16);
    u16x8 ahs = *(const u16x8*)(arow + k + hi * 16 + 8);
    u16x16 af = comb8(alo, ahs);
#pragma unroll
    for (int t = 0; t < 4; ++t) {
      const unsigned short* bp =
          Bt + (size_t)(colBase + t * 16 + lane16) * ldb + k + hi * 16;
      u16x16 bfr = *(const u16x16*)bp;
      c[t] = wmma_bf(af, bfr, c[t]);
    }
  }
}

// QKV GEMM: [8192 x 6144] = xb[8192x2048] @ WqkvT' + b; scatter into Qh/Kh/Vt (bf16)
__global__ void __launch_bounds__(128)
k_gemm_qkv(const unsigned short* __restrict__ xb,
           const unsigned short* __restrict__ wt,
           const float* __restrict__ bias,
           unsigned short* __restrict__ Qh,
           unsigned short* __restrict__ Kh,
           unsigned short* __restrict__ Vt) {
  const int wave   = threadIdx.x >> 5;
  const int lane16 = threadIdx.x & 15;
  const int hi     = (threadIdx.x >> 4) & 1;
  const int rowBase = blockIdx.y * 64 + wave * 16;
  const int colBase = blockIdx.x * 64;
  v8f c[4];
#pragma unroll
  for (int t = 0; t < 4; ++t) c[t] = (v8f){0, 0, 0, 0, 0, 0, 0, 0};
  gemm16x64(xb, DM, wt, DM, rowBase, colBase, DM, c);
#pragma unroll
  for (int t = 0; t < 4; ++t) {
    const int n  = colBase + t * 16 + lane16;
    const float bn = bias[n];
#pragma unroll
    for (int r = 0; r < 8; ++r) {
      const int m    = rowBase + r + 8 * hi;
      const int bIdx = m >> 11;          // /SEQ
      const int tpos = m & (SEQ - 1);
      const unsigned short bv = f2bf(c[t][r] + bn);
      if (n < DM) {
        const int hh = n >> 7, d = n & 127;
        Qh[(((size_t)(bIdx * NH + hh) * SEQ) + tpos) * HD + d] = bv;
      } else if (n < 2 * DM) {
        const int n2 = n - DM; const int hh = n2 >> 7, d = n2 & 127;
        Kh[(((size_t)(bIdx * NH + hh) * SEQ) + tpos) * HD + d] = bv;
      } else {
        const int n2 = n - 2 * DM; const int hh = n2 >> 7, d = n2 & 127;
        Vt[(((size_t)(bIdx * NH + hh) * HD) + d) * SEQ + tpos] = bv;  // transposed V
      }
    }
  }
}

// Projection GEMM: out[8192x2048] f32 = Oh @ WprojT' + b
__global__ void __launch_bounds__(128)
k_gemm_proj(const unsigned short* __restrict__ Oh,
            const unsigned short* __restrict__ wpt,
            const float* __restrict__ bias,
            float* __restrict__ out) {
  const int wave   = threadIdx.x >> 5;
  const int lane16 = threadIdx.x & 15;
  const int hi     = (threadIdx.x >> 4) & 1;
  const int rowBase = blockIdx.y * 64 + wave * 16;
  const int colBase = blockIdx.x * 64;
  v8f c[4];
#pragma unroll
  for (int t = 0; t < 4; ++t) c[t] = (v8f){0, 0, 0, 0, 0, 0, 0, 0};
  gemm16x64(Oh, DM, wpt, DM, rowBase, colBase, DM, c);
#pragma unroll
  for (int t = 0; t < 4; ++t) {
    const int n = colBase + t * 16 + lane16;
    const float bn = bias[n];
#pragma unroll
    for (int r = 0; r < 8; ++r) {
      const int m = rowBase + r + 8 * hi;
      out[(size_t)m * DM + n] = c[t][r] + bn;
    }
  }
}

// ---------------- flash attention: block = (b*h, 64 q rows), wave = 16 q rows ----------------
// K/V tiles are staged into LDS with double-buffered async-to-LDS DMA (ASYNCcnt).
__global__ void __launch_bounds__(128)
k_attn(const unsigned short* __restrict__ Qh,
       const unsigned short* __restrict__ Kh,
       const unsigned short* __restrict__ Vt,
       unsigned short* __restrict__ Oh) {
  __shared__ __align__(32) unsigned short kTile[2][KT * HD];   // 2 x 8 KB, [key][d]
  __shared__ __align__(32) unsigned short vTile[2][HD * KT];   // 2 x 8 KB, [d][key]
  __shared__ __align__(32) unsigned short pLds[4 * 16 * 32];   // 4 KB P staging
  const int tid    = threadIdx.x;
  const int lane   = tid & 31;
  const int lane16 = lane & 15;
  const int hi     = lane >> 4;
  const int wave   = tid >> 5;
  const int bh = blockIdx.x;
  const int b  = bh >> 4;
  const int h  = bh & 15;
  const int qb = blockIdx.y * 64 + wave * 16;

  const unsigned short* Qbh  = Qh + (size_t)bh * SEQ * HD;
  const unsigned short* Kbh  = Kh + (size_t)bh * SEQ * HD;
  const unsigned short* Vtbh = Vt + (size_t)bh * HD * SEQ;
  unsigned short* pl = pLds + wave * (16 * 32);

  // cooperative async stage of one 32-key K tile + V tile (8 x 16B per thread)
  auto issue_tile = [&](int j, int pb) {
    const int kb = j * KT;
    const unsigned short* ksrc = Kbh + (size_t)kb * HD;   // contiguous KT*HD block
#pragma unroll
    for (int i = 0; i < 4; ++i) {
      const int c = tid + 128 * i;                        // 16B chunk id, 0..511
      async_copy16(ksrc + c * 8, &kTile[pb][c * 8]);
    }
#pragma unroll
    for (int i = 0; i < 4; ++i) {
      const int c = tid + 128 * i;
      const int d = c >> 2, part = c & 3;                 // row d, 4 chunks per row
      async_copy16(Vtbh + (size_t)d * SEQ + kb + part * 8,
                   &vTile[pb][d * KT + part * 8]);
    }
  };

  // Q A-fragments for this wave's 16 rows, d = 0..127 in 4 chunks of 32
  u16x16 qf[4];
#pragma unroll
  for (int c = 0; c < 4; ++c) {
    const unsigned short* qp = Qbh + (size_t)(qb + lane16) * HD + c * 32 + hi * 16;
    qf[c] = comb8(*(const u16x8*)qp, *(const u16x8*)(qp + 8));
  }

  v8f o[8];
#pragma unroll
  for (int t = 0; t < 8; ++t) o[t] = (v8f){0, 0, 0, 0, 0, 0, 0, 0};
  float mrow[8], lrow[8];
#pragma unroll
  for (int r = 0; r < 8; ++r) { mrow[r] = -3.0e38f; lrow[r] = 0.f; }

  const float scale  = 0.08838834764831845f;   // 1/sqrt(128)
  const int   nTiles = (blockIdx.y + 1) * 2;   // uniform per block; mask handles ragged edge

  issue_tile(0, 0);
  for (int j = 0; j < nTiles; ++j) {
    const int pb = j & 1;
    const int kb = j * KT;
    if (j + 1 < nTiles) {
      issue_tile(j + 1, pb ^ 1);                       // prefetch next tile
      asm volatile("s_wait_asynccnt 8" ::: "memory");  // current tile's 8 loads done
    } else {
      asm volatile("s_wait_asynccnt 0" ::: "memory");
    }
    __syncthreads();   // tile visible to all waves

    // S = Q @ K^T : two 16-key sub-tiles, K-dim 128 = 4x32 (B-frags from LDS)
    v8f s0 = (v8f){0, 0, 0, 0, 0, 0, 0, 0};
    v8f s1 = (v8f){0, 0, 0, 0, 0, 0, 0, 0};
#pragma unroll
    for (int c = 0; c < 4; ++c) {
      const int d0 = c * 32 + hi * 16;
      u16x16 b0 = *(const u16x16*)(&kTile[pb][lane16 * HD + d0]);
      u16x16 b1 = *(const u16x16*)(&kTile[pb][(16 + lane16) * HD + d0]);
      s0 = wmma_bf(qf[c], b0, s0);
      s1 = wmma_bf(qf[c], b1, s1);
    }
    // online softmax over both sub-tiles; reductions across the 16-lane N group
    float p0a[8], p1a[8];
#pragma unroll
    for (int r = 0; r < 8; ++r) {
      const int qpos = qb + r + 8 * hi;
      const int k0 = kb + lane16;
      const int k1 = kb + 16 + lane16;
      float v0 = s0[r] * scale;
      float v1 = s1[r] * scale;
      const bool ok0 = (k0 <= qpos);
      const bool ok1 = (k1 <= qpos);
      float mx = fmaxf(ok0 ? v0 : -3.0e38f, ok1 ? v1 : -3.0e38f);
#pragma unroll
      for (int xm = 1; xm < 16; xm <<= 1) mx = fmaxf(mx, __shfl_xor(mx, xm, 32));
      float mnew = fmaxf(mrow[r], mx);
      float p0 = ok0 ? __expf(v0 - mnew) : 0.f;
      float p1 = ok1 ? __expf(v1 - mnew) : 0.f;
      float rs = p0 + p1;
#pragma unroll
      for (int xm = 1; xm < 16; xm <<= 1) rs += __shfl_xor(rs, xm, 32);
      float alpha = __expf(mrow[r] - mnew);
      lrow[r] = lrow[r] * alpha + rs;
      mrow[r] = mnew;
      p0a[r] = p0; p1a[r] = p1;
#pragma unroll
      for (int t = 0; t < 8; ++t) o[t][r] *= alpha;
    }
    // stage P (C-layout) into LDS row-major 16x32, reload as A-fragment
#pragma unroll
    for (int r = 0; r < 8; ++r) {
      const int m = r + 8 * hi;
      pl[m * 32 + lane16]      = f2bf(p0a[r]);
      pl[m * 32 + 16 + lane16] = f2bf(p1a[r]);
    }
    asm volatile("s_wait_dscnt 0" ::: "memory");   // DS ops are in-order per wave
    u16x8 plo = *(const u16x8*)(pl + lane16 * 32 + hi * 16);
    u16x8 phs = *(const u16x8*)(pl + lane16 * 32 + hi * 16 + 8);
    u16x16 pf = comb8(plo, phs);
    // O += P @ V : 8 d-tiles of 16, K-dim = 32 keys (B-frags from LDS V tile)
#pragma unroll
    for (int t = 0; t < 8; ++t) {
      u16x16 bv = *(const u16x16*)(&vTile[pb][(t * 16 + lane16) * KT + hi * 16]);
      o[t] = wmma_bf(pf, bv, o[t]);
    }
    __syncthreads();   // all reads of this buffer done before it is refilled
  }
  // normalize and merge heads into [B,T,C] bf16 for the projection GEMM
#pragma unroll
  for (int r = 0; r < 8; ++r) {
    const float inv = 1.0f / lrow[r];
    const int tpos = qb + r + 8 * hi;
    unsigned short* orow = Oh + ((size_t)(b * SEQ + tpos) * DM) + h * HD;
#pragma unroll
    for (int t = 0; t < 8; ++t) orow[t * 16 + lane16] = f2bf(o[t][r] * inv);
  }
}

// ---------------- host orchestration ----------------

extern "C" void kernel_launch(void* const* d_in, const int* in_sizes, int n_in,
                              void* d_out, int out_size, void* d_ws, size_t ws_size,
                              hipStream_t stream) {
  const float* x     = (const float*)d_in[0];
  const float* cosT  = (const float*)d_in[1];
  const float* sinT  = (const float*)d_in[2];
  const float* Wqkv  = (const float*)d_in[3];
  const float* bqkv  = (const float*)d_in[4];
  const float* Wproj = (const float*)d_in[5];
  const float* bproj = (const float*)d_in[6];
  float* out = (float*)d_out;

  char* ws = (char*)d_ws;
  size_t off = 0;
  auto take = [&](size_t bytes) -> char* {
    char* p = ws + off;
    off += (bytes + 255) & ~(size_t)255;
    return p;
  };
  unsigned short* xb     = (unsigned short*)take((size_t)MROWS * DM * 2);
  unsigned short* wqkvt  = (unsigned short*)take((size_t)N3 * DM * 2);
  unsigned short* wprojt = (unsigned short*)take((size_t)DM * DM * 2);
  unsigned short* Qh     = (unsigned short*)take((size_t)BHN * SEQ * HD * 2);
  unsigned short* Kh     = (unsigned short*)take((size_t)BHN * SEQ * HD * 2);
  unsigned short* Vt     = (unsigned short*)take((size_t)BHN * SEQ * HD * 2);
  unsigned short* Oh     = (unsigned short*)take((size_t)MROWS * DM * 2);
  (void)ws_size; (void)in_sizes; (void)n_in; (void)out_size;

  {
    int n = MROWS * DM;
    k_f32_to_bf16<<<n / 256, 256, 0, stream>>>(x, xb, n);
  }
  {
    int n = N3 * DM;
    k_transpose_w<<<n / 256, 256, 0, stream>>>(Wqkv, wqkvt, N3, n);
  }
  {
    int n = DM * DM;
    k_transpose_w<<<n / 256, 256, 0, stream>>>(Wproj, wprojt, DM, n);
  }
  {
    dim3 g(N3 / 64, MROWS / 64);
    k_gemm_qkv<<<g, 128, 0, stream>>>(xb, wqkvt, bqkv, Qh, Kh, Vt);
  }
  {
    int n = 2 * BHN * SEQ * (HD / 2);
    k_rope<<<n / 256, 256, 0, stream>>>(Qh, Kh, cosT, sinT, n);
  }
  {
    dim3 g(BHN, SEQ / 64);
    k_attn<<<g, 128, 0, stream>>>(Qh, Kh, Vt, Oh);
  }
  {
    dim3 g(DM / 64, MROWS / 64);
    k_gemm_proj<<<g, 128, 0, stream>>>(Oh, wprojt, bproj, out);
  }
}
